// RSSM_56573309223262
// MI455X (gfx1250) — matrix operand order repaired and checked by reading, not compiled, not run on hardware
//
#include <hip/hip_runtime.h>
#include <hip/hip_bf16.h>
#include <math.h>

// ---------------------------------------------------------------------------
// RSSM (Dreamer world-model step) for MI455X / gfx1250.
// All GEMMs run on v_wmma_f32_16x16x32_bf16 (fp32 accumulate), with the
// block-shared A tile double-buffered through LDS (ds_load_b128) and the
// weight stream prefetched one K-step ahead (global_prefetch_b8).
// Dims: B=256, T=64, O=2048, A=16, L=32, C=32, H=512, R=1024, LC=1024.
// ---------------------------------------------------------------------------

#define Bb   256
#define Tt   64
#define Oo   2048
#define Aa   16
#define Hh   512
#define Rr   1024
#define LCc  1024
#define G3R  3072          // 3*R
#define KX   1088          // A+LC=1040 padded to /64
#define KPJ  2048          // R+LC
#define KPO  3072          // R+O

#define LDS_STRIDE 72      // 64 + 8 pad (bf16 elems) -> conflict-free ds_load_b128
#define LDS_TILE   (32 * LDS_STRIDE)

typedef __attribute__((ext_vector_type(16))) __bf16 bf16x16;
typedef __attribute__((ext_vector_type(8)))  float  f32x8;
typedef unsigned short u16;
typedef unsigned long long u64;

union FragBF { uint4 u[2]; bf16x16 v; };

__device__ __forceinline__ u16 f2bf(float f) {
    union { float f; unsigned u; } x; x.f = f;
    unsigned u = x.u;
    unsigned r = (u + 0x7FFFu + ((u >> 16) & 1u)) >> 16;   // round-nearest-even
    return (u16)r;
}
__device__ __forceinline__ float bf2f(u16 h) {
    union { unsigned u; float f; } x; x.u = ((unsigned)h) << 16;
    return x.f;
}
__device__ __forceinline__ float hash_u01(u64 s) {
    s ^= s >> 33; s *= 0xff51afd7ed558ccdULL;
    s ^= s >> 33; s *= 0xc4ceb9fe1a85ec53ULL;
    s ^= s >> 33;
    return ((float)(unsigned)(s >> 40) + 0.5f) * (1.0f / 16777216.0f);
}

// ---------------------------------------------------------------------------
// GEMM:  out[M,N] = bf16A[M,K] * bf16W[N,K]^T + bias[N]   (optional relu)
// Block: 256 threads = 8 waves; block tile 32(M) x 256(N);
// wave tile 16(M) x 64(N) = 4 accumulators of v_wmma_f32_16x16x32_bf16.
// A tile (32x64 bf16) staged in LDS, double buffered; K-step 64.
// Grid:  (N/256, M/32).  Requires K % 64 == 0.
// ---------------------------------------------------------------------------
__global__ __launch_bounds__(256) void gemm_bf16_wmma(
    const u16* __restrict__ A, int lda,
    const u16* __restrict__ W, int ldw,
    const float* __restrict__ bias,
    float* __restrict__ outF, int ldc,
    u16* __restrict__ outB, int ldb,
    int K, int do_relu)
{
    __shared__ u16 smem[2 * LDS_TILE];

    const int tid  = threadIdx.x;
    const int wave = tid >> 5;
    const int lane = tid & 31;
    const int hl   = lane >> 4;     // half of wave
    const int ln   = lane & 15;
    const int wm   = wave & 1;
    const int wn   = wave >> 1;
    const int rowBase = blockIdx.y * 32  + wm * 16;
    const int colBase = blockIdx.x * 256 + wn * 64;

    f32x8 acc0 = {}, acc1 = {}, acc2 = {}, acc3 = {};

    // cooperative A tile load: thread -> (row = tid>>3, 8-elem chunk = tid&7)
    const int arow = tid >> 3;
    const int acol = (tid & 7) * 8;
    const u16* aglob = A + (size_t)(blockIdx.y * 32 + arow) * lda + acol;
    const int stoff  = arow * LDS_STRIDE + acol;

    // per-lane A fragment base inside LDS tile (row-major, padded stride)
    const int afrag = (wm * 16 + ln) * LDS_STRIDE + hl * 8;

    const u16* b0 = W + (size_t)(colBase + ln) * ldw + hl * 8;
    const u16* b1 = b0 + (size_t)16 * ldw;
    const u16* b2 = b0 + (size_t)32 * ldw;
    const u16* b3 = b0 + (size_t)48 * ldw;

    const int nk = K >> 6;

    // prologue: stage tile 0
    *(uint4*)(&smem[stoff]) = *(const uint4*)(aglob);
    __syncthreads();

    for (int i = 0; i < nk; ++i) {
        const int k0 = i << 6;
        const u16* cbuf = smem + (i & 1) * LDS_TILE;

        // fetch next A tile from global while computing on current one
        uint4 nxt;
        const bool more = (i + 1) < nk;
        if (more) {
            nxt = *(const uint4*)(aglob + k0 + 64);
            // prefetch next K-step of the weight stream into cache
            __builtin_prefetch(b0 + k0 + 64, 0, 3);
            __builtin_prefetch(b1 + k0 + 64, 0, 3);
            __builtin_prefetch(b2 + k0 + 64, 0, 3);
            __builtin_prefetch(b3 + k0 + 64, 0, 3);
        }

        // A fragments for both 32-wide K halves (ds_load_b128 x4)
        FragBF fa0, fa1;
        fa0.u[0] = *(const uint4*)(&cbuf[afrag]);
        fa0.u[1] = *(const uint4*)(&cbuf[afrag + 16]);
        fa1.u[0] = *(const uint4*)(&cbuf[afrag + 32]);
        fa1.u[1] = *(const uint4*)(&cbuf[afrag + 48]);

#pragma unroll
        for (int h2 = 0; h2 < 2; ++h2) {
            const int kk = k0 + h2 * 32;
            FragBF f0, f1, f2, f3;
            f0.u[0] = *(const uint4*)(b0 + kk); f0.u[1] = *(const uint4*)(b0 + kk + 16);
            f1.u[0] = *(const uint4*)(b1 + kk); f1.u[1] = *(const uint4*)(b1 + kk + 16);
            f2.u[0] = *(const uint4*)(b2 + kk); f2.u[1] = *(const uint4*)(b2 + kk + 16);
            f3.u[0] = *(const uint4*)(b3 + kk); f3.u[1] = *(const uint4*)(b3 + kk + 16);
            const bf16x16 av = h2 ? fa1.v : fa0.v;
            acc0 = __builtin_amdgcn_wmma_f32_16x16x32_bf16(false, av, false, f0.v, (short)0, acc0, false, false);
            acc1 = __builtin_amdgcn_wmma_f32_16x16x32_bf16(false, av, false, f1.v, (short)0, acc1, false, false);
            acc2 = __builtin_amdgcn_wmma_f32_16x16x32_bf16(false, av, false, f2.v, (short)0, acc2, false, false);
            acc3 = __builtin_amdgcn_wmma_f32_16x16x32_bf16(false, av, false, f3.v, (short)0, acc3, false, false);
        }

        if (more) *(uint4*)(&smem[((i + 1) & 1) * LDS_TILE + stoff]) = nxt;
        __syncthreads();
    }

    f32x8 accs[4] = { acc0, acc1, acc2, acc3 };
#pragma unroll
    for (int nt = 0; nt < 4; ++nt) {
        const int N = colBase + nt * 16 + ln;
        const float bi = bias[N];
        f32x8 a = accs[nt];
#pragma unroll
        for (int r = 0; r < 8; ++r) {
            const int M = rowBase + hl * 8 + r;   // D layout: VGPR r -> M = hl*8+r
            float v = a[r] + bi;
            if (do_relu) v = fmaxf(v, 0.0f);
            if (outF) outF[(size_t)M * ldc + N] = v;
            if (outB) outB[(size_t)M * ldb + N] = f2bf(v);
        }
    }
}

// --------------------------- pointwise kernels -----------------------------

__global__ void k_cvt_pad(const float* __restrict__ src, u16* __restrict__ dst,
                          int rows, int K, int Kpad) {
    int idx = blockIdx.x * blockDim.x + threadIdx.x;
    if (idx >= rows * Kpad) return;
    int r = idx / Kpad, k = idx - r * Kpad;
    dst[idx] = (k < K) ? f2bf(src[(size_t)r * K + k]) : (u16)0;
}

__global__ void k_init(u16* xbuf, u16* hbf, float* hf) {
    int idx = blockIdx.x * blockDim.x + threadIdx.x;
    if (idx < Bb * KX) xbuf[idx] = 0;
    if (idx < Bb * Rr) { hbf[idx] = 0; hf[idx] = 0.0f; }
}

__global__ void k_gru(const float* __restrict__ gx, const float* __restrict__ gh,
                      float* __restrict__ h, float* __restrict__ hid_out,
                      u16* __restrict__ h_bf, u16* __restrict__ post_in,
                      u16* __restrict__ joint) {
    int idx = blockIdx.x * blockDim.x + threadIdx.x;   // 256*1024
    int b = idx >> 10, j = idx & 1023;
    size_t gi = (size_t)b * G3R;
    float xr = gx[gi + j], xz = gx[gi + Rr + j], xn = gx[gi + 2 * Rr + j];
    float hr = gh[gi + j], hz = gh[gi + Rr + j], hn = gh[gi + 2 * Rr + j];
    float r = 1.0f / (1.0f + expf(-(xr + hr)));
    float z = 1.0f / (1.0f + expf(-(xz + hz)));
    float n = tanhf(xn + r * hn);
    float hv = (1.0f - z) * n + z * h[idx];
    h[idx] = hv;
    hid_out[(size_t)b * (Tt * Rr) + j] = hv;
    u16 hb = f2bf(hv);
    h_bf[idx] = hb;
    post_in[(size_t)b * KPO + j] = hb;
    joint[(size_t)b * KPJ + j] = hb;
}

__global__ void k_obs(const float* __restrict__ obs, u16* __restrict__ post_in, int t) {
    int idx = blockIdx.x * blockDim.x + threadIdx.x;   // 256*2048
    int b = idx >> 11, j = idx & 2047;
    post_in[(size_t)b * KPO + Rr + j] = f2bf(obs[((size_t)b * Tt + t) * Oo + j]);
}

__global__ void k_acta(const float* __restrict__ action, u16* __restrict__ xbuf, int t) {
    int idx = blockIdx.x * blockDim.x + threadIdx.x;   // 256*16
    int b = idx >> 4, a = idx & 15;
    xbuf[(size_t)b * KX + a] = f2bf(action[((size_t)b * Tt + t) * Aa + a]);
}

// Gumbel-max categorical sample over C=32 per (b, l); writes one-hot latent.
__global__ void k_sample(const float* __restrict__ qlog, float* __restrict__ latent_out,
                         u16* __restrict__ joint, u16* __restrict__ xbuf, int t) {
    int idx = blockIdx.x * blockDim.x + threadIdx.x;   // 256*32
    int b = idx >> 5, l = idx & 31;
    const float* q = qlog + (size_t)b * (Tt * LCc) + l * 32;
    int best = 0; float bestv = -1e30f;
#pragma unroll
    for (int c = 0; c < 32; ++c) {
        u64 key = ((u64)(unsigned)(t * 1000003 + idx)) * 0x9E3779B97F4A7C15ULL + (unsigned)(c * 2654435761u);
        float u = hash_u01(key);
        float g = -logf(-logf(u));
        float v = q[c] + g;
        if (v > bestv) { bestv = v; best = c; }
    }
    float* lo = latent_out + (size_t)b * (Tt * LCc) + l * 32;
    u16* jo = joint + (size_t)b * KPJ + Rr + l * 32;
    u16* xo = xbuf + (size_t)b * KX + Aa + l * 32;
#pragma unroll
    for (int c = 0; c < 32; ++c) {
        float v = (c == best) ? 1.0f : 0.0f;
        u16 bv = (c == best) ? (u16)0x3F80 : (u16)0;   // bf16 1.0
        lo[c] = v; jo[c] = bv; xo[c] = bv;
    }
}

// reward head second layer: N=1 -> plain dot product per batch row
__global__ void k_rew2(const u16* __restrict__ h1r, const float* __restrict__ w2,
                       const float* __restrict__ b2, float* __restrict__ out) {
    int b = threadIdx.x;           // 256 threads, 1 block
    float s = b2[0];
    for (int k = 0; k < Hh; ++k) s += bf2f(h1r[(size_t)b * Hh + k]) * w2[k];
    out[(size_t)b * Tt] = s;
}

// ---------------------------------------------------------------------------

extern "C" void kernel_launch(void* const* d_in, const int* in_sizes, int n_in,
                              void* d_out, int out_size, void* d_ws, size_t ws_size,
                              hipStream_t stream) {
    const float* obs      = (const float*)d_in[0];
    const float* action   = (const float*)d_in[1];
    const float* w_ih     = (const float*)d_in[2];
    const float* w_hh     = (const float*)d_in[3];
    const float* b_ih     = (const float*)d_in[4];
    const float* b_hh     = (const float*)d_in[5];
    const float* prior_w1 = (const float*)d_in[6];
    const float* prior_b1 = (const float*)d_in[7];
    const float* prior_w2 = (const float*)d_in[8];
    const float* prior_b2 = (const float*)d_in[9];
    const float* post_w1  = (const float*)d_in[10];
    const float* post_b1  = (const float*)d_in[11];
    const float* post_w2  = (const float*)d_in[12];
    const float* post_b2  = (const float*)d_in[13];
    const float* dec_w1   = (const float*)d_in[14];
    const float* dec_b1   = (const float*)d_in[15];
    const float* dec_w2   = (const float*)d_in[16];
    const float* dec_b2   = (const float*)d_in[17];
    const float* rew_w1   = (const float*)d_in[18];
    const float* rew_b1   = (const float*)d_in[19];
    const float* rew_w2   = (const float*)d_in[20];
    const float* rew_b2   = (const float*)d_in[21];

    float* out = (float*)d_out;
    const size_t latent_base = 0;
    const size_t hidden_base = (size_t)Bb * Tt * LCc;
    const size_t plog_base   = 2 * hidden_base;
    const size_t qlog_base   = 3 * hidden_base;
    const size_t recon_base  = 4 * hidden_base;
    const size_t rew_base    = recon_base + (size_t)Bb * Tt * Oo;

    // workspace bump allocator
    char* cur = (char*)d_ws;
    auto alloc = [&](size_t bytes) -> void* {
        void* p = (void*)cur;
        cur += (bytes + 255) & ~(size_t)255;
        return p;
    };
    u16* wih_bf = (u16*)alloc((size_t)G3R * KX  * 2);
    u16* whh_bf = (u16*)alloc((size_t)G3R * Rr  * 2);
    u16* pw1    = (u16*)alloc((size_t)Hh  * Rr  * 2);
    u16* pw2    = (u16*)alloc((size_t)LCc * Hh  * 2);
    u16* qw1    = (u16*)alloc((size_t)Hh  * KPO * 2);
    u16* qw2    = (u16*)alloc((size_t)LCc * Hh  * 2);
    u16* dw1    = (u16*)alloc((size_t)Hh  * KPJ * 2);
    u16* dw2    = (u16*)alloc((size_t)Oo  * Hh  * 2);
    u16* rw1    = (u16*)alloc((size_t)Hh  * KPJ * 2);
    u16* xbuf   = (u16*)alloc((size_t)Bb * KX  * 2);
    u16* hbf    = (u16*)alloc((size_t)Bb * Rr  * 2);
    float* hf   = (float*)alloc((size_t)Bb * Rr * 4);
    u16* postin = (u16*)alloc((size_t)Bb * KPO * 2);
    u16* joint  = (u16*)alloc((size_t)Bb * KPJ * 2);
    u16* h1p    = (u16*)alloc((size_t)Bb * Hh * 2);
    u16* h1q    = (u16*)alloc((size_t)Bb * Hh * 2);
    u16* h1d    = (u16*)alloc((size_t)Bb * Hh * 2);
    u16* h1r    = (u16*)alloc((size_t)Bb * Hh * 2);
    float* gxb  = (float*)alloc((size_t)Bb * G3R * 4);
    float* ghb  = (float*)alloc((size_t)Bb * G3R * 4);

    auto cvt = [&](const float* s, u16* d, int rows, int K, int Kpad) {
        int n = rows * Kpad;
        k_cvt_pad<<<(n + 255) / 256, 256, 0, stream>>>(s, d, rows, K, Kpad);
    };
    // weight fp32 -> bf16 (once per launch; deterministic)
    cvt(w_ih,     wih_bf, G3R, Aa + LCc, KX);
    cvt(w_hh,     whh_bf, G3R, Rr,  Rr);
    cvt(prior_w1, pw1,    Hh,  Rr,  Rr);
    cvt(prior_w2, pw2,    LCc, Hh,  Hh);
    cvt(post_w1,  qw1,    Hh,  KPO, KPO);
    cvt(post_w2,  qw2,    LCc, Hh,  Hh);
    cvt(dec_w1,   dw1,    Hh,  KPJ, KPJ);
    cvt(dec_w2,   dw2,    Oo,  Hh,  Hh);
    cvt(rew_w1,   rw1,    Hh,  KPJ, KPJ);

    {
        int n = Bb * KX;
        k_init<<<(n + 255) / 256, 256, 0, stream>>>(xbuf, hbf, hf);
    }

    auto gemm = [&](const u16* A, int lda, const u16* W, int ldw, const float* bias,
                    float* outF, int ldc, u16* outB, int ldb,
                    int M, int N, int K, int relu) {
        gemm_bf16_wmma<<<dim3(N / 256, M / 32), 256, 0, stream>>>(
            A, lda, W, ldw, bias, outF, ldc, outB, ldb, K, relu);
    };

    for (int t = 0; t < Tt; ++t) {
        // gate pre-activations: gx = x @ w_ih^T + b_ih ; gh = h @ w_hh^T + b_hh
        gemm(xbuf, KX, wih_bf, KX, b_ih, gxb, G3R, nullptr, 0, Bb, G3R, KX, 0);
        gemm(hbf,  Rr, whh_bf, Rr, b_hh, ghb, G3R, nullptr, 0, Bb, G3R, Rr, 0);
        // GRU pointwise update; fan h out (fp32 + bf16 copies into A-buffers)
        k_gru<<<(Bb * Rr) / 256, 256, 0, stream>>>(
            gxb, ghb, hf, out + hidden_base + (size_t)t * Rr, hbf, postin, joint);
        // o_t into post_in[:, R:R+O]
        k_obs<<<(Bb * Oo) / 256, 256, 0, stream>>>(obs, postin, t);
        // prior MLP -> plog
        gemm(hbf, Rr, pw1, Rr, prior_b1, nullptr, 0, h1p, Hh, Bb, Hh, Rr, 1);
        gemm(h1p, Hh, pw2, Hh, prior_b2, out + plog_base + (size_t)t * LCc, Tt * LCc,
             nullptr, 0, Bb, LCc, Hh, 0);
        // posterior MLP -> qlog
        gemm(postin, KPO, qw1, KPO, post_b1, nullptr, 0, h1q, Hh, Bb, Hh, KPO, 1);
        gemm(h1q, Hh, qw2, Hh, post_b2, out + qlog_base + (size_t)t * LCc, Tt * LCc,
             nullptr, 0, Bb, LCc, Hh, 0);
        // sample one-hot latent; feed joint and next-step x
        k_sample<<<(Bb * 32) / 256, 256, 0, stream>>>(
            out + qlog_base + (size_t)t * LCc, out + latent_base + (size_t)t * LCc,
            joint, xbuf, t);
        // a_t -> x for step t+1
        k_acta<<<(Bb * Aa) / 256, 256, 0, stream>>>(action, xbuf, t);
        // decoder -> recon
        gemm(joint, KPJ, dw1, KPJ, dec_b1, nullptr, 0, h1d, Hh, Bb, Hh, KPJ, 1);
        gemm(h1d, Hh, dw2, Hh, dec_b2, out + recon_base + (size_t)t * Oo, Tt * Oo,
             nullptr, 0, Bb, Oo, Hh, 0);
        // reward head
        gemm(joint, KPJ, rw1, KPJ, rew_b1, nullptr, 0, h1r, Hh, Bb, Hh, KPJ, 1);
        k_rew2<<<1, 256, 0, stream>>>(h1r, rew_w2, rew_b2, out + rew_base + t);
    }
}
